// net_mgnn_84524956385830
// MI455X (gfx1250) — matrix operand order repaired and checked by reading, not compiled, not run on hardware
//
#include <hip/hip_runtime.h>
#include <math.h>

#define N_NODES 50000
#define N_EDGES 800000
#define CH      128
#define ALPHA   0.5f
#define EPS_ATT 1e-5f

typedef __attribute__((ext_vector_type(16))) _Float16 v16h;
typedef __attribute__((ext_vector_type(8)))  _Float16 v8h;
typedef __attribute__((ext_vector_type(8)))  float    v8f;

__device__ __forceinline__ v8f wmma_f16(v16h a, v16h b, v8f c) {
  // D = A(16x32 f16) * B(32x16 f16) + C(16x16 f32)
  return __builtin_amdgcn_wmma_f32_16x16x32_f16(false, a, false, b, (short)0, c,
                                                false, false);
}

__device__ __forceinline__ void cvt8(v16h& r, int base, float4 u, float4 v) {
  r[base + 0] = (_Float16)u.x; r[base + 1] = (_Float16)u.y;
  r[base + 2] = (_Float16)u.z; r[base + 3] = (_Float16)u.w;
  r[base + 4] = (_Float16)v.x; r[base + 5] = (_Float16)v.y;
  r[base + 6] = (_Float16)v.z; r[base + 7] = (_Float16)v.w;
}

// A-fragment (16x32) from row-major fp32; src points at tile origin (m0,k0).
// ISA layout: element e of lane -> K = 16*(e>>3) + 8*(lane>=16) + (e&7).
__device__ __forceinline__ v16h load_a_f32v(const float* src, int ld, int lane) {
  int m = lane & 15, half = (lane >> 4) & 1;
  const float* p = src + m * ld + half * 8;
  float4 u0 = *(const float4*)(p + 0);
  float4 u1 = *(const float4*)(p + 4);
  float4 u2 = *(const float4*)(p + 16);
  float4 u3 = *(const float4*)(p + 20);
  v16h r;
  cvt8(r, 0, u0, u1);
  cvt8(r, 8, u2, u3);
  return r;
}

// A-fragment blended on the fly: sup = (1-ALPHA)*agg + ALPHA*x0
__device__ __forceinline__ v16h load_a_blend(const float* pa, const float* px,
                                             int ld, int lane) {
  int m = lane & 15, half = (lane >> 4) & 1;
  const float* a = pa + m * ld + half * 8;
  const float* b = px + m * ld + half * 8;
  float4 a0 = *(const float4*)(a + 0),  a1 = *(const float4*)(a + 4);
  float4 a2 = *(const float4*)(a + 16), a3 = *(const float4*)(a + 20);
  float4 b0 = *(const float4*)(b + 0),  b1 = *(const float4*)(b + 4);
  float4 b2 = *(const float4*)(b + 16), b3 = *(const float4*)(b + 20);
  float4 s0, s1, s2, s3;
  s0.x = (1.0f-ALPHA)*a0.x + ALPHA*b0.x; s0.y = (1.0f-ALPHA)*a0.y + ALPHA*b0.y;
  s0.z = (1.0f-ALPHA)*a0.z + ALPHA*b0.z; s0.w = (1.0f-ALPHA)*a0.w + ALPHA*b0.w;
  s1.x = (1.0f-ALPHA)*a1.x + ALPHA*b1.x; s1.y = (1.0f-ALPHA)*a1.y + ALPHA*b1.y;
  s1.z = (1.0f-ALPHA)*a1.z + ALPHA*b1.z; s1.w = (1.0f-ALPHA)*a1.w + ALPHA*b1.w;
  s2.x = (1.0f-ALPHA)*a2.x + ALPHA*b2.x; s2.y = (1.0f-ALPHA)*a2.y + ALPHA*b2.y;
  s2.z = (1.0f-ALPHA)*a2.z + ALPHA*b2.z; s2.w = (1.0f-ALPHA)*a2.w + ALPHA*b2.w;
  s3.x = (1.0f-ALPHA)*a3.x + ALPHA*b3.x; s3.y = (1.0f-ALPHA)*a3.y + ALPHA*b3.y;
  s3.z = (1.0f-ALPHA)*a3.z + ALPHA*b3.z; s3.w = (1.0f-ALPHA)*a3.w + ALPHA*b3.w;
  v16h r;
  cvt8(r, 0, s0, s1);
  cvt8(r, 8, s2, s3);
  return r;
}

// A-fragment from f16 LDS buffer (two 16B vector loads)
__device__ __forceinline__ v16h load_a_f16v(const _Float16* src, int ld, int lane) {
  int m = lane & 15, half = (lane >> 4) & 1;
  const _Float16* p = src + m * ld + half * 8;
  v8h lo = *(const v8h*)(p);
  v8h hi = *(const v8h*)(p + 16);
  return __builtin_shufflevector(lo, hi, 0, 1, 2, 3, 4, 5, 6, 7,
                                 8, 9, 10, 11, 12, 13, 14, 15);
}

// B-fragment from pre-packed fragment-major f16 buffer: 32B contiguous per lane.
__device__ __forceinline__ v16h load_b_pk(const _Float16* frag, int tileIdx, int lane) {
  const v8h* p = (const v8h*)(frag + ((long)tileIdx * 32 + lane) * 16);
  v8h lo = p[0];
  v8h hi = p[1];
  return __builtin_shufflevector(lo, hi, 0, 1, 2, 3, 4, 5, 6, 7,
                                 8, 9, 10, 11, 12, 13, 14, 15);
}

// ---- pack one weight matrix (K x Ncols fp32) into WMMA B-fragment layout ----
// blockIdx.x = tile t = kt*NT + nt ; 32 threads ; lane layout matches load_b_pk.
__global__ void k_pack(const float* __restrict__ W, int Ncols, int nvalid, int NT,
                       _Float16* __restrict__ frag) {
  int t = blockIdx.x;
  int kt = t / NT, nt = t % NT;
  int lane = threadIdx.x & 31;
  int n = lane & 15, half = (lane >> 4) & 1;
  _Float16* dst = frag + ((long)t * 32 + lane) * 16;
  if (n < nvalid) {
    int ncol = nt * 16 + n;
#pragma unroll
    for (int e = 0; e < 16; ++e)
      dst[e] = (_Float16)W[(long)(kt * 32 + 16 * (e >> 3) + half * 8 + (e & 7)) * Ncols + ncol];
  } else {
#pragma unroll
    for (int e = 0; e < 16; ++e) dst[e] = (_Float16)0.0f;
  }
}

// ---------------- fused 3-layer MLP -> logits (N x 8) ----------------
__global__ void k_mlp(const float* __restrict__ x,
                      const _Float16* __restrict__ pk1, const float* __restrict__ bp1,
                      const _Float16* __restrict__ pk2, const float* __restrict__ bp2,
                      const _Float16* __restrict__ pk3, const float* __restrict__ bp3,
                      float* __restrict__ logits) {
  __shared__ alignas(16) _Float16 h1s[2][16 * 512];
  __shared__ alignas(16) _Float16 h2s[2][16 * 64];
  int wave = threadIdx.x >> 5, lane = threadIdx.x & 31;
  int tile = blockIdx.x * 2 + wave;
  if (tile * 16 >= N_NODES) return;
  const float* xa = x + (long)tile * 16 * CH;
  int nn = lane & 15, half = (lane >> 4) & 1;

  v16h af[16];
  // stage 1: h1 = relu(x @ wp1 + bp1)  (16x512), K=128
#pragma unroll
  for (int ks = 0; ks < 4; ++ks) af[ks] = load_a_f32v(xa + ks * 32, CH, lane);
  for (int nt = 0; nt < 32; ++nt) {
    v8f c = {};
#pragma unroll
    for (int ks = 0; ks < 4; ++ks)
      c = wmma_f16(af[ks], load_b_pk(pk1, ks * 32 + nt, lane), c);
    float bias = bp1[nt * 16 + nn];
#pragma unroll
    for (int v = 0; v < 8; ++v)
      h1s[wave][(v + 8 * half) * 512 + nt * 16 + nn] =
          (_Float16)fmaxf(c[v] + bias, 0.0f);
  }
  // stage 2: h2 = relu(h1 @ wp2 + bp2)  (16x64), K=512
#pragma unroll
  for (int ks = 0; ks < 16; ++ks)
    af[ks] = load_a_f16v(&h1s[wave][ks * 32], 512, lane);
  for (int nt = 0; nt < 4; ++nt) {
    v8f c = {};
#pragma unroll
    for (int ks = 0; ks < 16; ++ks)
      c = wmma_f16(af[ks], load_b_pk(pk2, ks * 4 + nt, lane), c);
    float bias = bp2[nt * 16 + nn];
#pragma unroll
    for (int v = 0; v < 8; ++v)
      h2s[wave][(v + 8 * half) * 64 + nt * 16 + nn] =
          (_Float16)fmaxf(c[v] + bias, 0.0f);
  }
  // stage 3: logits = h2 @ wp3 + bp3  (16x8), K=64
#pragma unroll
  for (int ks = 0; ks < 2; ++ks)
    af[ks] = load_a_f16v(&h2s[wave][ks * 32], 64, lane);
  v8f c = {};
#pragma unroll
  for (int ks = 0; ks < 2; ++ks)
    c = wmma_f16(af[ks], load_b_pk(pk3, ks, lane), c);
  if (nn < 8) {
    float bias = bp3[nn];
#pragma unroll
    for (int v = 0; v < 8; ++v)
      logits[((long)tile * 16 + v + 8 * half) * 8 + nn] = c[v] + bias;
  }
}

// ------------- per-edge bilinear weight + sum / sumsq reduction -------------
__global__ void k_edge_ew(const float* __restrict__ logits, const int* __restrict__ erow,
                          const int* __restrict__ ecol, const float* __restrict__ pm0,
                          float* __restrict__ ew_raw, float* __restrict__ red) {
  __shared__ float Ps[64];
  __shared__ float s1[256], s2[256];
  int t = threadIdx.x;
  if (t < 64) Ps[t] = fmaxf(2.0f * pm0[t], 0.0f);
  __syncthreads();
  int e = blockIdx.x * 256 + t;
  float val = 0.0f, vsq = 0.0f;
  if (e < N_EDGES) {
    int r = erow[e], c = ecol[e];
    const float4* lp = (const float4*)logits;
    float4 s0 = lp[(long)r * 2], sA = lp[(long)r * 2 + 1];
    float4 e0 = lp[(long)c * 2], eA = lp[(long)c * 2 + 1];
    float s[8] = {s0.x, s0.y, s0.z, s0.w, sA.x, sA.y, sA.z, sA.w};
    float en[8] = {e0.x, e0.y, e0.z, e0.w, eA.x, eA.y, eA.z, eA.w};
    float acc = 0.0f;
#pragma unroll
    for (int j = 0; j < 8; ++j) {
      float tj = 0.0f;
#pragma unroll
      for (int k = 0; k < 8; ++k) tj += en[k] * Ps[k * 8 + j];
      acc += s[j] * tj;
    }
    ew_raw[e] = acc;
    val = acc; vsq = acc * acc;
  }
  s1[t] = val; s2[t] = vsq;
  __syncthreads();
  for (int off = 128; off > 0; off >>= 1) {
    if (t < off) { s1[t] += s1[t + off]; s2[t] += s2[t + off]; }
    __syncthreads();
  }
  if (t == 0) { atomicAdd(&red[0], s1[0]); atomicAdd(&red[1], s2[0]); }
}

__global__ void k_deg(const int* __restrict__ ecol, float* __restrict__ deg) {
  int e = blockIdx.x * 256 + threadIdx.x;
  if (e < N_EDGES) atomicAdd(&deg[ecol[e]], 1.0f);
}

__global__ void k_dis(float* __restrict__ deg) {
  int i = blockIdx.x * 256 + threadIdx.x;
  if (i < N_NODES) { float d = deg[i]; deg[i] = (d > 0.0f) ? rsqrtf(d) : 0.0f; }
}

__global__ void k_ewfin(const float* __restrict__ red, float* __restrict__ ew) {
  int e = blockIdx.x * 256 + threadIdx.x;
  if (e >= N_EDGES) return;
  float s1 = red[0], s2 = red[1];
  float Ef = (float)N_EDGES;
  float mean = s1 / Ef;
  float var = (s2 - s1 * s1 / Ef) / (Ef - 1.0f);  // unbiased (ddof=1)
  float scale = sqrtf(1e-4f / var);
  ew[e] = (ew[e] - mean) * scale + 1.0f;
}

// ---------------- initial projection x @ w_init + b_init ----------------
__global__ void k_proj(const float* __restrict__ x, const _Float16* __restrict__ pkw,
                       const float* __restrict__ b_init, float* __restrict__ x0,
                       float* __restrict__ xcur) {
  int wave = threadIdx.x >> 5, lane = threadIdx.x & 31;
  int tile = blockIdx.x * 4 + wave;
  if (tile * 16 >= N_NODES) return;
  const float* xa = x + (long)tile * 16 * CH;
  int nn = lane & 15, half = (lane >> 4) & 1;
  v16h af[4];
#pragma unroll
  for (int ks = 0; ks < 4; ++ks) af[ks] = load_a_f32v(xa + ks * 32, CH, lane);
  for (int nt = 0; nt < 8; ++nt) {
    v8f c = {};
#pragma unroll
    for (int ks = 0; ks < 4; ++ks)
      c = wmma_f16(af[ks], load_b_pk(pkw, ks * 8 + nt, lane), c);
    float bias = b_init[nt * 16 + nn];
#pragma unroll
    for (int v = 0; v < 8; ++v) {
      long gi = ((long)tile * 16 + v + 8 * half) * CH + nt * 16 + nn;
      float r = c[v] + bias;
      x0[gi] = r;
      xcur[gi] = r;
    }
  }
}

// ------- edge attention (WMMA) + metric gate -> effective edge weight -------
__global__ void k_edge_att(const float* __restrict__ x0, const int* __restrict__ erow,
                           const int* __restrict__ ecol, const _Float16* __restrict__ pkatt,
                           const float* __restrict__ b_att, const float* __restrict__ v_att,
                           const float* __restrict__ dis, const float* __restrict__ ew,
                           float* __restrict__ w_e) {
  __shared__ alignas(16) _Float16 cat[4][16 * 256];  // concat(xi,xj) per wave, f16
  __shared__ float nrm[4][16];
  __shared__ float alog[4][16];
  int wave = threadIdx.x >> 5, lane = threadIdx.x & 31;
  int tile = blockIdx.x * 4 + wave;
  if (tile * 16 >= N_EDGES) return;
  int e0 = tile * 16;
  const float4* xv = (const float4*)x0;
  for (int i = 0; i < 16; ++i) {
    int r = erow[e0 + i], c = ecol[e0 + i];
    float4 a4 = xv[(long)r * 32 + lane];
    float4 b4 = xv[(long)c * 32 + lane];
    int o = i * 256 + lane * 4;
    cat[wave][o + 0] = (_Float16)a4.x; cat[wave][o + 1] = (_Float16)a4.y;
    cat[wave][o + 2] = (_Float16)a4.z; cat[wave][o + 3] = (_Float16)a4.w;
    cat[wave][o + 128 + 0] = (_Float16)b4.x; cat[wave][o + 128 + 1] = (_Float16)b4.y;
    cat[wave][o + 128 + 2] = (_Float16)b4.z; cat[wave][o + 128 + 3] = (_Float16)b4.w;
    float dx = a4.x - b4.x, dy = a4.y - b4.y, dz = a4.z - b4.z, dw = a4.w - b4.w;
    float d2 = dx * dx + dy * dy + dz * dz + dw * dw;
#pragma unroll
    for (int off = 16; off > 0; off >>= 1) d2 += __shfl_xor(d2, off, 32);
    if (lane == 0) nrm[wave][i] = sqrtf(d2);
  }
  if (lane < 16) alog[wave][lane] = 0.0f;
  int nn = lane & 15, half = (lane >> 4) & 1;
  v16h af[8];
#pragma unroll
  for (int ks = 0; ks < 8; ++ks)
    af[ks] = load_a_f16v(&cat[wave][ks * 32], 256, lane);
  for (int nt = 0; nt < 8; ++nt) {
    v8f c = {};
#pragma unroll
    for (int ks = 0; ks < 8; ++ks)
      c = wmma_f16(af[ks], load_b_pk(pkatt, ks * 8 + nt, lane), c);
    float bias = b_att[nt * 16 + nn];
    float vat = v_att[nt * 16 + nn];
#pragma unroll
    for (int v = 0; v < 8; ++v) {
      float h = fmaxf(c[v] + bias, 0.0f);            // relu(att_h)
      atomicAdd(&alog[wave][v + 8 * half], h * vat); // dot with v_att
    }
  }
  if (lane < 16) {
    int e = e0 + lane;
    float a = 1.0f / (1.0f + expf(-alog[wave][lane]));  // sigmoid
    float em = (1.0f - a) / (1.0f + a + EPS_ATT) * nrm[wave][lane];
    w_e[e] = dis[erow[e]] * dis[ecol[e]] * ew[e] * expf(-em);
  }
}

// ------------- weighted scatter-aggregate: agg += w_e * x[col] -------------
// one wave per edge (e is wave-uniform -> scalar loads for edge metadata)
__global__ void k_scatter(const float* __restrict__ xcur, const int* __restrict__ erow,
                          const int* __restrict__ ecol, const float* __restrict__ w_e,
                          float* __restrict__ agg) {
  int lane = threadIdx.x & 31;
  long e = (long)blockIdx.x * 8 + (threadIdx.x >> 5);
  if (e >= N_EDGES) return;
  float w = w_e[e];
  int r = erow[e], c = ecol[e];
  const float4* xv = (const float4*)xcur;
  float4 xc = xv[(long)c * 32 + lane];
  float* dst = agg + (long)r * CH + lane * 4;
  atomicAdd(dst + 0, w * xc.x);
  atomicAdd(dst + 1, w * xc.y);
  atomicAdd(dst + 2, w * xc.z);
  atomicAdd(dst + 3, w * xc.w);
}

// ---- GCNII layer: x = relu((1-b)*sup + b*(sup@wc)), sup=(1-a)agg+a*x0 ----
__global__ void k_layer(const float* __restrict__ agg, const float* __restrict__ x0,
                        const _Float16* __restrict__ pkwc, float beta,
                        float* __restrict__ xcur) {
  int wave = threadIdx.x >> 5, lane = threadIdx.x & 31;
  int tile = blockIdx.x * 4 + wave;
  if (tile * 16 >= N_NODES) return;
  long base = (long)tile * 16 * CH;
  int nn = lane & 15, half = (lane >> 4) & 1;
  v16h af[4];
#pragma unroll
  for (int ks = 0; ks < 4; ++ks)
    af[ks] = load_a_blend(agg + base + ks * 32, x0 + base + ks * 32, CH, lane);
  for (int nt = 0; nt < 8; ++nt) {
    v8f c = {};
#pragma unroll
    for (int ks = 0; ks < 4; ++ks)
      c = wmma_f16(af[ks], load_b_pk(pkwc, ks * 8 + nt, lane), c);
#pragma unroll
    for (int v = 0; v < 8; ++v) {
      long idx = base + (long)(v + 8 * half) * CH + nt * 16 + nn;
      float s = (1.0f - ALPHA) * agg[idx] + ALPHA * x0[idx];  // fp32 sup
      float r = (1.0f - beta) * s + beta * c[v];
      xcur[idx] = fmaxf(r, 0.0f);
    }
  }
}

// ---------------- final: out = x @ w_final + b_final ----------------
__global__ void k_final(const float* __restrict__ xcur, const _Float16* __restrict__ pkw,
                        const float* __restrict__ b_final, float* __restrict__ out) {
  int wave = threadIdx.x >> 5, lane = threadIdx.x & 31;
  int tile = blockIdx.x * 4 + wave;
  if (tile * 16 >= N_NODES) return;
  const float* xa = xcur + (long)tile * 16 * CH;
  int nn = lane & 15, half = (lane >> 4) & 1;
  v16h af[4];
#pragma unroll
  for (int ks = 0; ks < 4; ++ks) af[ks] = load_a_f32v(xa + ks * 32, CH, lane);
  v8f c = {};
#pragma unroll
  for (int ks = 0; ks < 4; ++ks)
    c = wmma_f16(af[ks], load_b_pk(pkw, ks, lane), c);
  if (nn < 8) {
    float bias = b_final[nn];
#pragma unroll
    for (int v = 0; v < 8; ++v)
      out[((long)tile * 16 + v + 8 * half) * 8 + nn] = c[v] + bias;
  }
}

extern "C" void kernel_launch(void* const* d_in, const int* in_sizes, int n_in,
                              void* d_out, int out_size, void* d_ws, size_t ws_size,
                              hipStream_t stream) {
  (void)in_sizes; (void)n_in; (void)out_size; (void)ws_size;
  const float* x       = (const float*)d_in[0];
  const int*   eidx    = (const int*)d_in[1];
  const float* w_init  = (const float*)d_in[2];
  const float* b_init  = (const float*)d_in[3];
  const float* w_att   = (const float*)d_in[4];
  const float* b_att   = (const float*)d_in[5];
  const float* v_att   = (const float*)d_in[6];
  const float* wp1     = (const float*)d_in[7];
  const float* bp1     = (const float*)d_in[8];
  const float* wp2     = (const float*)d_in[9];
  const float* bp2     = (const float*)d_in[10];
  const float* wp3     = (const float*)d_in[11];
  const float* bp3     = (const float*)d_in[12];
  const float* pm0     = (const float*)d_in[13];
  const float* wc0     = (const float*)d_in[14];
  const float* wc1     = (const float*)d_in[15];
  const float* w_final = (const float*)d_in[16];
  const float* b_final = (const float*)d_in[17];
  const int* erow = eidx;
  const int* ecol = eidx + N_EDGES;
  float* out = (float*)d_out;

  float* ws = (float*)d_ws;
  size_t off = 0;
  float* logits = ws + off; off += (size_t)N_NODES * 8;
  float* ew     = ws + off; off += N_EDGES;
  float* red    = ws + off; off += 64;
  float* deg    = ws + off; off += N_NODES + 64;     // becomes dis in place
  float* w_e    = ws + off; off += N_EDGES;
  float* x0     = ws + off; off += (size_t)N_NODES * CH;
  float* xcur   = ws + off; off += (size_t)N_NODES * CH;
  float* agg    = ws + off; off += (size_t)N_NODES * CH;

  // packed f16 weight fragments (fragment-major, 512 f16 per 16x16x32 tile)
  _Float16* pkbase = (_Float16*)(ws + off);
  size_t po = 0;
  _Float16* pk_wp1  = pkbase + po; po += (size_t)(4 * 32) * 512;
  _Float16* pk_wp2  = pkbase + po; po += (size_t)(16 * 4) * 512;
  _Float16* pk_wp3  = pkbase + po; po += (size_t)(2 * 1) * 512;
  _Float16* pk_wini = pkbase + po; po += (size_t)(4 * 8) * 512;
  _Float16* pk_watt = pkbase + po; po += (size_t)(8 * 8) * 512;
  _Float16* pk_wc0  = pkbase + po; po += (size_t)(4 * 8) * 512;
  _Float16* pk_wc1  = pkbase + po; po += (size_t)(4 * 8) * 512;
  _Float16* pk_wfin = pkbase + po; po += (size_t)(4 * 1) * 512;

  hipMemsetAsync(red, 0, 2 * sizeof(float), stream);
  hipMemsetAsync(deg, 0, (size_t)N_NODES * sizeof(float), stream);

  // pack all weight matrices into WMMA fragment layout (f16)
  k_pack<<<4 * 32, 32, 0, stream>>>(wp1,     512, 16, 32, pk_wp1);
  k_pack<<<16 * 4, 32, 0, stream>>>(wp2,      64, 16,  4, pk_wp2);
  k_pack<<<2 * 1,  32, 0, stream>>>(wp3,       8,  8,  1, pk_wp3);
  k_pack<<<4 * 8,  32, 0, stream>>>(w_init,  128, 16,  8, pk_wini);
  k_pack<<<8 * 8,  32, 0, stream>>>(w_att,   128, 16,  8, pk_watt);
  k_pack<<<4 * 8,  32, 0, stream>>>(wc0,     128, 16,  8, pk_wc0);
  k_pack<<<4 * 8,  32, 0, stream>>>(wc1,     128, 16,  8, pk_wc1);
  k_pack<<<4 * 1,  32, 0, stream>>>(w_final,   8,  8,  1, pk_wfin);

  int tilesN = (N_NODES + 15) / 16;   // 3125
  int tilesE = (N_EDGES + 15) / 16;   // 50000

  k_mlp<<<(tilesN + 1) / 2, 64, 0, stream>>>(x, pk_wp1, bp1, pk_wp2, bp2, pk_wp3, bp3,
                                             logits);
  k_edge_ew<<<(N_EDGES + 255) / 256, 256, 0, stream>>>(logits, erow, ecol, pm0, ew, red);
  k_deg<<<(N_EDGES + 255) / 256, 256, 0, stream>>>(ecol, deg);
  k_dis<<<(N_NODES + 255) / 256, 256, 0, stream>>>(deg);
  k_ewfin<<<(N_EDGES + 255) / 256, 256, 0, stream>>>(red, ew);
  k_proj<<<(tilesN + 3) / 4, 128, 0, stream>>>(x, pk_wini, b_init, x0, xcur);
  k_edge_att<<<(tilesE + 3) / 4, 128, 0, stream>>>(x0, erow, ecol, pk_watt, b_att, v_att,
                                                   deg, ew, w_e);

  const float beta1 = 0.4054651081081644f;   // log(0.5/1 + 1)
  const float beta2 = 0.22314355131420976f;  // log(0.5/2 + 1)
  for (int l = 0; l < 2; ++l) {
    hipMemsetAsync(agg, 0, (size_t)N_NODES * CH * sizeof(float), stream);
    k_scatter<<<(N_EDGES + 7) / 8, 256, 0, stream>>>(xcur, erow, ecol, w_e, agg);
    k_layer<<<(tilesN + 3) / 4, 128, 0, stream>>>(agg, x0, (l == 0) ? pk_wc0 : pk_wc1,
                                                  (l == 0) ? beta1 : beta2, xcur);
  }
  k_final<<<(tilesN + 3) / 4, 128, 0, stream>>>(xcur, pk_wfin, b_final, out);
}
